// LinkPredictor_1881195676360
// MI455X (gfx1250) — matrix operand order repaired and checked by reading, not compiled, not run on hardware
//
#include <hip/hip_runtime.h>

typedef __attribute__((ext_vector_type(2)))  float  v2f;
typedef __attribute__((ext_vector_type(8)))  float  v8f;
typedef __attribute__((ext_vector_type(16))) __bf16 v16bf;

namespace {
constexpr int NN = 50000;   // nodes
constexpr int D  = 128;     // feature dim
constexpr int NE = 600000;  // edges
constexpr int NP = 200000;  // predicted edges (out_edges is [2, NP])
constexpr int NS = 300000;  // selected rows
constexpr int UD = 256;     // dense units

constexpr int ROW_TILES = NN / 16;  // 3125 (exact)
constexpr int SEL_TILES = NS / 16;  // 18750 (exact)

__device__ inline unsigned bf16_rne(float f) {
  unsigned x = __float_as_uint(f);
  unsigned r = x + 0x7FFFu + ((x >> 16) & 1u);
  return r >> 16;
}

__device__ inline v16bf as_v16bf(const unsigned u[8]) {
  union { unsigned u[8]; v16bf v; } un;
#pragma unroll
  for (int i = 0; i < 8; ++i) un.u[i] = u[i];
  return un.v;
}
}  // namespace

// ---------------------------------------------------------------- utilities
__global__ void zero_f32(float* __restrict__ p, int n) {
  int i = blockIdx.x * blockDim.x + threadIdx.x;
  if (i < n) p[i] = 0.0f;
}

__global__ void count_degs(const int* __restrict__ src, const int* __restrict__ dst,
                           float* __restrict__ dego, float* __restrict__ degi) {
  int e = blockIdx.x * blockDim.x + threadIdx.x;
  if (e < NE) {
    atomicAdd(&dego[src[e]], 1.0f);
    atomicAdd(&degi[dst[e]], 1.0f);
  }
}

__global__ void finalize_degs(float* __restrict__ g, int n) {
  int i = blockIdx.x * blockDim.x + threadIdx.x;
  if (i < n) {
    float d = g[i];
    g[i] = rsqrtf(d < 1.0f ? 1.0f : d);
  }
}

// Pack W[128][128] into (k,k+1)-pair-major float2: Wp[k2*D + n] = {W[2k2][n], W[2k2+1][n]}
__global__ void pack_w_pairs(const float* __restrict__ W, float* __restrict__ Wp) {
  int t = blockIdx.x * blockDim.x + threadIdx.x;  // over (D/2)*D = 8192
  if (t >= (D / 2) * D) return;
  int k2 = t >> 7, n = t & (D - 1);
  Wp[2 * t + 0] = W[(2 * k2 + 0) * D + n];
  Wp[2 * t + 1] = W[(2 * k2 + 1) * D + n];
}

// Pack dense_W[128][256] into bf16 pairs: Bp[k2*UD + n] = bf16(W[2k2][n]) | bf16(W[2k2+1][n])<<16
__global__ void pack_dense_bf16(const float* __restrict__ W, unsigned* __restrict__ Bp) {
  int t = blockIdx.x * blockDim.x + threadIdx.x;  // over (D/2)*UD = 16384
  if (t >= (D / 2) * UD) return;
  int k2 = t >> 8, n = t & (UD - 1);
  unsigned lo = bf16_rne(W[(2 * k2 + 0) * UD + n]);
  unsigned hi = bf16_rne(W[(2 * k2 + 1) * UD + n]);
  Bp[t] = lo | (hi << 16);
}

// ------------------------------------------------- conv GEMM: Y = (X .* rinv) @ W
// One wave computes a 16x128 output tile via 32 x (8 cols) V_WMMA_F32_16X16X4_F32.
// Layouts (ISA 7.12.2): A 16x4 f32: lane m=lane&15, V0=K(2*half), V1=K(2*half+1)
//                       B 4x16 f32: lane n=lane&15, V0=K(2*half), V1=K(2*half+1)
//                       C/D 16x16:  lane n=lane&15, Vv = row v + 8*half
__global__ __launch_bounds__(256)
void conv_gemm_f32(const float* __restrict__ X, const float* __restrict__ rinv,
                   const float* __restrict__ Wp, float* __restrict__ Y) {
  int wave = (blockIdx.x * blockDim.x + threadIdx.x) >> 5;
  int lane = threadIdx.x & 31;
  if (wave >= ROW_TILES) return;  // wave-uniform exit: EXEC all-1s for WMMA
  int lm = lane & 15, half = lane >> 4;
  int row0 = wave * 16;
  int r = row0 + lm;
  float rs = rinv[r];
  const float* Xr = X + (size_t)r * D;

  v8f acc[8];
#pragma unroll
  for (int t = 0; t < 8; ++t)
#pragma unroll
    for (int v = 0; v < 8; ++v) acc[t][v] = 0.0f;

  for (int k = 0; k < D; k += 4) {
    int kk = k + 2 * half;
    v2f a = *(const v2f*)(Xr + kk);
    a[0] *= rs;
    a[1] *= rs;
    const float* wbase = Wp + (size_t)(kk >> 1) * D * 2;
#pragma unroll
    for (int t = 0; t < 8; ++t) {
      v2f b = *(const v2f*)(wbase + (t * 16 + lm) * 2);
      acc[t] = __builtin_amdgcn_wmma_f32_16x16x4_f32(
          false, a, false, b, (short)0, acc[t], false, false);
    }
  }
#pragma unroll
  for (int t = 0; t < 8; ++t)
#pragma unroll
    for (int v = 0; v < 8; ++v)
      Y[(size_t)(row0 + v + 8 * half) * D + t * 16 + lm] = acc[t][v];
}

// ------------------------------------------------- edge scatter: H[dst] += tmp[src]
// One wave per edge: lane covers 4 consecutive features (coalesced 512B row moves,
// src row + dst accumulator both L2-resident: 25.6 MB each << 192 MB L2).
__global__ __launch_bounds__(256)
void scatter_edges(const float* __restrict__ T, const int* __restrict__ src,
                   const int* __restrict__ dst, float* __restrict__ H) {
  int wave = (blockIdx.x * blockDim.x + threadIdx.x) >> 5;
  int lane = threadIdx.x & 31;
  if (wave >= NE) return;
  int s = src[wave], d = dst[wave];
  const float4 v = *(const float4*)(T + (size_t)s * D + lane * 4);
  float* hp = H + (size_t)d * D + lane * 4;
  atomicAdd(hp + 0, v.x);
  atomicAdd(hp + 1, v.y);
  atomicAdd(hp + 2, v.z);
  atomicAdd(hp + 3, v.w);
}

__global__ void conv_epilogue(float* __restrict__ H, const float* __restrict__ rinv_in,
                              const float* __restrict__ b) {
  int i = blockIdx.x * blockDim.x + threadIdx.x;
  if (i < NN * D) {
    int n = i >> 7, f = i & (D - 1);
    H[i] = H[i] * rinv_in[n] + b[f];
  }
}

// ------------------------------------------------- fused head:
// rows = set_idx -> out_edges -> h2 gather; hidden = relu(x@dense_W+b) via
// V_WMMA_F32_16X16X32_BF16 (16 col tiles x 4 K-steps); out layer folded in as
// per-lane partial logits, reduced with a 16-lane shfl_xor tree; softmax; store.
__global__ __launch_bounds__(256)
void head_fused(const float* __restrict__ H, const int* __restrict__ out_edges,
                const int* __restrict__ set_idx, const unsigned* __restrict__ Bp,
                const float* __restrict__ dense_b, const float* __restrict__ out_W,
                const float* __restrict__ out_b, float* __restrict__ out) {
  int wave = (blockIdx.x * blockDim.x + threadIdx.x) >> 5;
  int lane = threadIdx.x & 31;
  if (wave >= SEL_TILES) return;  // wave-uniform exit
  int lm = lane & 15, half = lane >> 4;
  int row0 = wave * 16;
  int i = row0 + lm;
  // concat([h[oe0], h[oe1]])[g] == h[out_edges[g]] because out_edges is [2,NP] flat
  int g = set_idx[i];
  int node = out_edges[g];
  const float* x = H + (size_t)node * D;

  // A fragments (bf16, 16x32 per K-step), all four K-steps kept in registers.
  // ISA layout: v<4 -> K = k0 + 2v + 8*half ; v>=4 -> K = k0 + 16 + 2(v-4) + 8*half
  unsigned afr[4][8];
#pragma unroll
  for (int q = 0; q < 4; ++q) {
    int k0 = q * 32;
#pragma unroll
    for (int v = 0; v < 8; ++v) {
      int K = k0 + ((v < 4) ? (2 * v + 8 * half) : (16 + 2 * (v - 4) + 8 * half));
      v2f xv = *(const v2f*)(x + K);
      afr[q][v] = bf16_rne(xv[0]) | (bf16_rne(xv[1]) << 16);
    }
  }

  float pl0[8], pl1[8];
#pragma unroll
  for (int v = 0; v < 8; ++v) { pl0[v] = 0.0f; pl1[v] = 0.0f; }

  for (int t = 0; t < UD / 16; ++t) {  // 16 hidden column tiles
    int n0 = t * 16;
    float bias = dense_b[n0 + lm];
    v8f c;
#pragma unroll
    for (int v = 0; v < 8; ++v) c[v] = bias;
#pragma unroll
    for (int q = 0; q < 4; ++q) {
      // B 32x16 bf16: lane n=lm, v<4 -> K pair (k0 + 8*half + 2v, +1); v>=4 -> K pair (k0 + 16 + 8*half + 2(v-4), +1)
      int kp = q * 16 + 4 * half;  // pair-index base
      unsigned bfr[8];
#pragma unroll
      for (int v = 0; v < 8; ++v) bfr[v] = Bp[(size_t)(kp + v + ((v >= 4) ? 4 : 0)) * UD + n0 + lm];
      c = __builtin_amdgcn_wmma_f32_16x16x32_bf16(
          false, as_v16bf(afr[q]), false, as_v16bf(bfr), (short)0, c, false, false);
    }
    float2 ow = *(const float2*)(out_W + (n0 + lm) * 2);
#pragma unroll
    for (int v = 0; v < 8; ++v) {
      float h = c[v] > 0.0f ? c[v] : 0.0f;  // ReLU
      pl0[v] += h * ow.x;                    // fold 256->2 projection
      pl1[v] += h * ow.y;
    }
  }

  // Reduce partial logits across the 16 lanes of each half-wave (xor < 16 never
  // crosses the half boundary).
#pragma unroll
  for (int off = 1; off < 16; off <<= 1) {
#pragma unroll
    for (int v = 0; v < 8; ++v) {
      pl0[v] += __shfl_xor(pl0[v], off, 32);
      pl1[v] += __shfl_xor(pl1[v], off, 32);
    }
  }

  if (lm == 0) {  // lane 0 -> rows 0..7, lane 16 -> rows 8..15
    float ob0 = out_b[0], ob1 = out_b[1];
#pragma unroll
    for (int v = 0; v < 8; ++v) {
      int m = v + 8 * half;
      float l0 = pl0[v] + ob0, l1 = pl1[v] + ob1;
      float mx = fmaxf(l0, l1);
      float e0 = __expf(l0 - mx), e1 = __expf(l1 - mx);
      float inv = 1.0f / (e0 + e1);
      out[(size_t)(row0 + m) * 2 + 0] = e0 * inv;
      out[(size_t)(row0 + m) * 2 + 1] = e1 * inv;
    }
  }
}

// ---------------------------------------------------------------- launcher
extern "C" void kernel_launch(void* const* d_in, const int* in_sizes, int n_in,
                              void* d_out, int out_size, void* d_ws, size_t ws_size,
                              hipStream_t stream) {
  const float* node_state = (const float*)d_in[0];
  const int*   edge_index = (const int*)d_in[1];  // [2, NE] flat
  const int*   out_edges  = (const int*)d_in[2];  // [2, NP] flat
  const int*   set_idx    = (const int*)d_in[3];
  const float* W1 = (const float*)d_in[4];
  const float* b1 = (const float*)d_in[5];
  const float* W2 = (const float*)d_in[6];
  const float* b2 = (const float*)d_in[7];
  const float* dense_W = (const float*)d_in[8];
  const float* dense_b = (const float*)d_in[9];
  const float* out_W = (const float*)d_in[10];
  const float* out_b = (const float*)d_in[11];
  float* out = (float*)d_out;

  const int* src = edge_index;
  const int* dst = edge_index + NE;

  // workspace layout (floats)
  float* ws = (float*)d_ws;
  float* rinvO = ws;                       // 50048
  float* rinvI = ws + 50048;               // 50048
  float* tmp   = ws + 100096;              // 6,400,000
  float* hbuf  = ws + 6500096;             // 6,400,000
  float* wp    = ws + 12900096;            // 16384 (pair-packed conv W)
  unsigned* bp = (unsigned*)(ws + 12916480);  // 16384 u32 (bf16-packed dense_W)

  auto cdiv = [](int a, int b) { return (a + b - 1) / b; };
  const int T = 256;

  // degrees -> rsqrt(max(deg,1))
  zero_f32<<<cdiv(2 * 50048, T), T, 0, stream>>>(rinvO, 2 * 50048);
  count_degs<<<cdiv(NE, T), T, 0, stream>>>(src, dst, rinvO, rinvI);
  finalize_degs<<<cdiv(2 * 50048, T), T, 0, stream>>>(rinvO, 2 * 50048);

  // GraphConv 1
  pack_w_pairs<<<cdiv((D / 2) * D, T), T, 0, stream>>>(W1, wp);
  conv_gemm_f32<<<cdiv(ROW_TILES * 32, T), T, 0, stream>>>(node_state, rinvO, wp, tmp);
  zero_f32<<<cdiv(NN * D, T), T, 0, stream>>>(hbuf, NN * D);
  scatter_edges<<<cdiv(NE * 32, T), T, 0, stream>>>(tmp, src, dst, hbuf);
  conv_epilogue<<<cdiv(NN * D, T), T, 0, stream>>>(hbuf, rinvI, b1);

  // GraphConv 2 (ping-pong: hbuf -> tmp -> hbuf)
  pack_w_pairs<<<cdiv((D / 2) * D, T), T, 0, stream>>>(W2, wp);
  conv_gemm_f32<<<cdiv(ROW_TILES * 32, T), T, 0, stream>>>(hbuf, rinvO, wp, tmp);
  zero_f32<<<cdiv(NN * D, T), T, 0, stream>>>(hbuf, NN * D);
  scatter_edges<<<cdiv(NE * 32, T), T, 0, stream>>>(tmp, src, dst, hbuf);
  conv_epilogue<<<cdiv(NN * D, T), T, 0, stream>>>(hbuf, rinvI, b2);

  // fused gather -> dense(bf16 WMMA) -> ReLU -> out -> softmax
  pack_dense_bf16<<<cdiv((D / 2) * UD, T), T, 0, stream>>>(dense_W, bp);
  head_fused<<<cdiv(SEL_TILES * 32, T), T, 0, stream>>>(
      hbuf, out_edges, set_idx, bp, dense_b, out_W, out_b, out);
}
